// Compressor_24180665876754
// MI455X (gfx1250) — compile-verified, matrix operand order checked by reading, not run on hardware
//
#include <hip/hip_runtime.h>

// ---------------------------------------------------------------------------
// Problem constants (match reference)
// ---------------------------------------------------------------------------
#define B_   2
#define L_   2048
#define D_   1024
#define H_   16
#define HD_  64
#define HID_ 4096
#define NL_  2
#define M_   (B_ * L_)          // 4096 token rows

// ---------------------------------------------------------------------------
// CDNA5 WMMA types / helpers (wave32, 16x16x32 bf16 -> f32)
// ---------------------------------------------------------------------------
typedef __attribute__((ext_vector_type(16))) __bf16 v16bf;
typedef __attribute__((ext_vector_type(8)))  float  v8f;
typedef __attribute__((ext_vector_type(4)))  int    i4v;
typedef __attribute__((ext_vector_type(4)))  float  f4v;
typedef __attribute__((ext_vector_type(4)))  unsigned int u32x4;
typedef __attribute__((ext_vector_type(8)))  int    i32x8;
typedef __attribute__((ext_vector_type(4)))  int    i32x4;

__device__ __forceinline__ unsigned short f2bf(float f) {
  unsigned int u = __float_as_uint(f);
  u += 0x7fffu + ((u >> 16) & 1u);              // round-to-nearest-even
  return (unsigned short)(u >> 16);
}
__device__ __forceinline__ float bf2f(unsigned short h) {
  return __uint_as_float(((unsigned int)h) << 16);
}

// Load a 16-element bf16 fragment as two contiguous 16B chunks (p, p+16 elems).
// Matches the ISA 16-bit A/B layouts: lane L<16 holds K=0..7 & 16..23,
// lane L+16 holds K=8..15 & 24..31 (caller passes the half*8 offset).
__device__ __forceinline__ v16bf ldfrag(const unsigned short* p) {
  union { i4v q[2]; v16bf v; } u;
  u.q[0] = *(const i4v*)(p);
  u.q[1] = *(const i4v*)(p + 16);
  return u.v;
}

__device__ __forceinline__ v8f wmma_bf16(v16bf a, v16bf b, v8f c) {
  return __builtin_amdgcn_wmma_f32_16x16x32_bf16(
      /*neg_a=*/false, a, /*neg_b=*/false, b,
      /*c_mod=*/(short)0, c, /*reuse_a=*/false, /*reuse_b=*/false);
}

__device__ __forceinline__ float silu_f(float x) { return x / (1.f + __expf(-x)); }

__device__ __forceinline__ float rmax16(float v) {
#pragma unroll
  for (int m = 1; m < 16; m <<= 1) v = fmaxf(v, __shfl_xor(v, m, 32));
  return v;
}
__device__ __forceinline__ float rsum16(float v) {
#pragma unroll
  for (int m = 1; m < 16; m <<= 1) v += __shfl_xor(v, m, 32);
  return v;
}

// ---------------------------------------------------------------------------
// Tensor Data Mover: DMA a 128x32 bf16 tile (row stride K elems) from global
// memory into LDS. Descriptor per CDNA5 ISA 8.3/8.4 (2-D tensor, groups 2/3
// zero). Issued by one wave; completion tracked with TENSORcnt.
// ---------------------------------------------------------------------------
__device__ __forceinline__ unsigned ldsoff(const void* p) {
  // flat LDS addresses are {SHARED_BASE[63:32], byte offset} -> low 32 bits
  return (unsigned)(unsigned long long)p;
}

__device__ __forceinline__ void tdm_load_tile_a(unsigned lds_addr,
                                                const unsigned short* gptr, int K) {
  unsigned long long ga = (unsigned long long)gptr;
  u32x4 g0;
  g0[0] = 1u;                                   // count=1 (valid user D#)
  g0[1] = lds_addr;                             // LDS byte address
  g0[2] = (unsigned)ga;                         // global_addr[31:0]
  g0[3] = (unsigned)(ga >> 32) | (2u << 30);    // global_addr[56:32] | type=2
  i32x8 g1;
  g1[0] = 0x00010000;                           // workgroup_mask=0, data_size=1 (2B)
  g1[1] = (K & 0xFFFF) << 16;                   // tensor_dim0[15:0]
  g1[2] = ((K >> 16) & 0xFFFF) | (M_ << 16);    // tensor_dim0[31:16] | tensor_dim1[15:0]
  g1[3] = (32 << 16);                           // tensor_dim1[31:16]=0 | tile_dim0=32
  g1[4] = 128;                                  // tile_dim1=128, tile_dim2=0
  g1[5] = K;                                    // tensor_dim0_stride[31:0]
  g1[6] = 0;                                    // stride hi / dim1_stride lo
  g1[7] = 0;
  i32x4 g2 = {0, 0, 0, 0};
  i32x4 g3 = {0, 0, 0, 0};
#if defined(__clang_major__) && (__clang_major__ >= 23)
  i32x8 g4 = {0, 0, 0, 0, 0, 0, 0, 0};
  __builtin_amdgcn_tensor_load_to_lds(g0, g1, g2, g3, g4, 0);
#else
  __builtin_amdgcn_tensor_load_to_lds(g0, g1, g2, g3, 0);
#endif
}

// ---------------------------------------------------------------------------
// Utility: vectorized f32 copy (x -> residual stream in ws; inputs immutable)
// ---------------------------------------------------------------------------
__global__ __launch_bounds__(256) void copy_f32(const float* __restrict__ src,
                                                float* __restrict__ dst, int n4) {
  int i = blockIdx.x * 256 + threadIdx.x;
  if (i < n4) ((f4v*)dst)[i] = ((const f4v*)src)[i];
}

// ---------------------------------------------------------------------------
// RMSNorm: one 256-thread block per token row (D=1024 -> 4 elems/thread).
// Writes bf16 (GEMM input); optionally also fp32 (final norm -> d_out).
// ---------------------------------------------------------------------------
template <bool FOUT>
__global__ __launch_bounds__(256) void rmsnorm_kernel(const float* __restrict__ x,
                                                      const float* __restrict__ w,
                                                      unsigned short* __restrict__ hbf,
                                                      float* __restrict__ xout) {
  __shared__ float red[256];
  const int row = blockIdx.x, tid = threadIdx.x;
  const float* xr = x + (size_t)row * D_;
  float v[4];
  float s = 0.f;
#pragma unroll
  for (int i = 0; i < 4; i++) { v[i] = xr[tid + i * 256]; s += v[i] * v[i]; }
  red[tid] = s;
  __syncthreads();
  for (int off = 128; off > 0; off >>= 1) {
    if (tid < off) red[tid] += red[tid + off];
    __syncthreads();
  }
  const float r = rsqrtf(red[0] / (float)D_ + 1e-5f);
#pragma unroll
  for (int i = 0; i < 4; i++) {
    int c = tid + i * 256;
    float o = v[i] * r * w[c];
    hbf[(size_t)row * D_ + c] = f2bf(o);
    if (FOUT) xout[(size_t)row * D_ + c] = o;
  }
}

// ---------------------------------------------------------------------------
// GEMM: C(M,N) = A(M,K)bf16 @ W(K,N)f32 (converted to bf16 on LDS stage).
// 256 threads = 8 waves. Block tile 128x128, K-step 32.
// A tile (raw bf16) is DMA'd by the Tensor Data Mover into double-buffered
// LDS: wave 0 issues tensor_load_to_lds for step i+1 right after the barrier,
// everyone executes s_wait_tensorcnt 0 at the top of the next iteration
// (no-op for non-issuing waves), so the DMA overlaps the 8 WMMAs of step i.
// B tile needs fp32->bf16 convert + [n][k] transpose, so it stays on the
// VALU path (also double-buffered). Next-step W lines are prefetched.
// Wave grid 4(M) x 2(N): each wave owns a 32x64 tile -> 2 A-frags x 4 B-frags
// -> 8 v_wmma_f32_16x16x32_bf16 per K-step per wave (64 f32 accum VGPRs).
// All dims (M=4096, N in {1024,4096}, K in {1024,4096}) divide the tiles.
// Epilogues:
//   0: Cf = c                       1: Cf = c + aux (residual, may alias Cf)
//   3: Cb = bf16(silu(c))           4: Cb = bf16(c * bf2f(Cb)) in-place gate mul
//   5: Cb = bf16(silu(c + bias[col]))
// ---------------------------------------------------------------------------
template <int EPI>
__global__ __launch_bounds__(256) void gemm_kernel(const unsigned short* __restrict__ A,
                                                   const float* __restrict__ W,
                                                   int N, int K,
                                                   float* __restrict__ Cf,
                                                   unsigned short* __restrict__ Cb,
                                                   const float* __restrict__ aux,
                                                   const float* __restrict__ bias) {
  __shared__ unsigned short lA[2][128 * 32];   // [m][k] bf16, TDM-filled, 2x8KB
  __shared__ unsigned short lB[2][128 * 32];   // [n][k] bf16 (transposed), 2x8KB
  const int tid  = threadIdx.x;
  const int m0   = blockIdx.x * 128, n0 = blockIdx.y * 128;
  const int wave = tid >> 5, lane = tid & 31, ln = lane & 15, half = lane >> 4;
  const int wm   = wave & 3;      // wave row tile  (32 rows)
  const int wn   = wave >> 2;     // wave col tile  (64 cols)

  v8f c[2][4];
#pragma unroll
  for (int mt = 0; mt < 2; mt++)
#pragma unroll
    for (int nt = 0; nt < 4; nt++)
#pragma unroll
      for (int r = 0; r < 8; r++) c[mt][nt][r] = 0.f;

  // B tile: 32x128 = 4096 f32 / 256 thr = 16 each (four f4v), coalesced along n
  const int lb_k = tid >> 3, lb_n = (tid & 7) * 16;
  const unsigned short* Abase = A + (size_t)m0 * K;

  // prologue: DMA first A tile
  if (wave == 0) tdm_load_tile_a(ldsoff(&lA[0][0]), Abase, K);

  int ibuf = 0;
  for (int k0 = 0; k0 < K; k0 += 32, ibuf ^= 1) {
    // A tile for this step has landed (TENSORcnt is per-wave; no-op elsewhere)
    __builtin_amdgcn_s_wait_tensorcnt(0);

    // stage B tile (fp32 -> bf16, [n][k]) into this step's buffer
    const float* gw = &W[(size_t)(k0 + lb_k) * N + n0 + lb_n];
#pragma unroll
    for (int q = 0; q < 4; q++) {
      f4v w4 = *(const f4v*)(gw + q * 4);
#pragma unroll
      for (int j = 0; j < 4; j++)
        lB[ibuf][(lb_n + q * 4 + j) * 32 + lb_k] = f2bf(w4[j]);
    }
    // prefetch next K-step of W (speculative; global_prefetch_b8)
    __builtin_prefetch(gw + (size_t)32 * N, 0, 1);
    __syncthreads();

    // kick off the DMA for the next A tile into the other buffer; it overlaps
    // the WMMA work below
    if (wave == 0 && k0 + 32 < K)
      tdm_load_tile_a(ldsoff(&lA[ibuf ^ 1][0]), Abase + k0 + 32, K);

    v16bf af[2], bfr[4];
#pragma unroll
    for (int mt = 0; mt < 2; mt++)
      af[mt] = ldfrag(&lA[ibuf][(wm * 32 + mt * 16 + ln) * 32 + half * 8]);
#pragma unroll
    for (int nt = 0; nt < 4; nt++)
      bfr[nt] = ldfrag(&lB[ibuf][(wn * 64 + nt * 16 + ln) * 32 + half * 8]);
#pragma unroll
    for (int mt = 0; mt < 2; mt++)
#pragma unroll
      for (int nt = 0; nt < 4; nt++)
        c[mt][nt] = wmma_bf16(af[mt], bfr[nt], c[mt][nt]);
  }

#pragma unroll
  for (int mt = 0; mt < 2; mt++)
#pragma unroll
    for (int nt = 0; nt < 4; nt++)
#pragma unroll
      for (int r = 0; r < 8; r++) {
        int row = m0 + wm * 32 + mt * 16 + r + 8 * half; // C layout: VGPR r -> M=r / r+8
        int col = n0 + wn * 64 + nt * 16 + ln;
        size_t idx = (size_t)row * N + col;
        float cv = c[mt][nt][r];
        if (EPI == 0)      Cf[idx] = cv;
        else if (EPI == 1) Cf[idx] = cv + aux[idx];
        else if (EPI == 3) Cb[idx] = f2bf(silu_f(cv));
        else if (EPI == 4) { float g = bf2f(Cb[idx]); Cb[idx] = f2bf(cv * g); }
        else if (EPI == 5) Cb[idx] = f2bf(silu_f(cv + bias[col]));
      }
}

// ---------------------------------------------------------------------------
// RoPE + head layout pack: (B*L, D) f32 -> bf16 [b,h,l,hd], optional 1/sqrt(hd)
// ---------------------------------------------------------------------------
__global__ __launch_bounds__(256) void rope_pack_kernel(const float* __restrict__ src,
                                                        const float* __restrict__ cosp,
                                                        const float* __restrict__ sinp,
                                                        unsigned short* __restrict__ dst,
                                                        float scale) {
  int gid = blockIdx.x * 256 + threadIdx.x;       // B*L*H*32 threads
  int p = gid & 31;
  int h = (gid >> 5) & (H_ - 1);
  int l = (gid >> 9) & (L_ - 1);
  int b = gid >> 20;
  size_t srow = (size_t)(b * L_ + l) * D_ + h * HD_;
  float c0 = src[srow + p], c1 = src[srow + p + 32];
  float co  = cosp[l * HD_ + p],  co2 = cosp[l * HD_ + p + 32];
  float si  = sinp[l * HD_ + p],  si2 = sinp[l * HD_ + p + 32];
  float o0 = (c0 * co  - c1 * si)  * scale;       // rot_half = [-q2, q1]
  float o1 = (c1 * co2 + c0 * si2) * scale;
  size_t drow = ((size_t)(b * H_ + h) * L_ + l) * HD_;
  dst[drow + p]      = f2bf(o0);
  dst[drow + p + 32] = f2bf(o1);
}

// V transpose pack: (B*L, D) f32 -> bf16 [b,h,hd,L] so PV B-fragments are
// contiguous along the key dimension.
__global__ __launch_bounds__(256) void vtrans_kernel(const float* __restrict__ src,
                                                     unsigned short* __restrict__ dst) {
  int gid = blockIdx.x * 256 + threadIdx.x;       // B*L*D threads
  int c = gid & (D_ - 1);
  int l = (gid >> 10) & (L_ - 1);
  int b = gid >> 21;
  int hd = c & 63, h = c >> 6;
  dst[((size_t)(b * H_ + h) * HD_ + hd) * L_ + l] = f2bf(src[gid]);
}

// ---------------------------------------------------------------------------
// Flash attention: one wave per (b,h, 16-query block). S=QK^T and O=PV on
// WMMA; online softmax on the f32 C-tile layout (row = r + 8*(lane>=16),
// col = lane&15) with shfl_xor reductions inside each 16-lane half.
// ---------------------------------------------------------------------------
__global__ __launch_bounds__(32) void attn_kernel(const unsigned short* __restrict__ qbf,
                                                  const unsigned short* __restrict__ kbf,
                                                  const unsigned short* __restrict__ vtbf,
                                                  unsigned short* __restrict__ obf) {
  __shared__ unsigned short lds_p[16 * 32];       // P tile relayout C -> A
  const int qblk = blockIdx.x, bh = blockIdx.y;
  const int b = bh >> 4, h = bh & (H_ - 1);
  const int lane = threadIdx.x, ln = lane & 15, half = lane >> 4;
  const int q0 = qblk * 16;
  const unsigned short* qb = qbf + (size_t)bh * L_ * HD_;
  const unsigned short* kb = kbf + (size_t)bh * L_ * HD_;
  const unsigned short* vb = vtbf + (size_t)bh * HD_ * L_;

  // Q A-fragments for head-dim chunks [0,32) and [32,64)
  v16bf aq0 = ldfrag(qb + (size_t)(q0 + ln) * HD_ + half * 8);
  v16bf aq1 = ldfrag(qb + (size_t)(q0 + ln) * HD_ + 32 + half * 8);

  v8f o[4];
  float m[8], lsum[8];
#pragma unroll
  for (int nt = 0; nt < 4; nt++)
#pragma unroll
    for (int r = 0; r < 8; r++) o[nt][r] = 0.f;
#pragma unroll
  for (int r = 0; r < 8; r++) { m[r] = -1e30f; lsum[r] = 0.f; }

  for (int jb = 0; jb < q0 + 16; jb += 32) {      // causal: keys < q0+16
    // prefetch next key block's K rows / V columns while we compute this one
    __builtin_prefetch(kb + (size_t)(jb + 32 + ln) * HD_, 0, 1);
    __builtin_prefetch(vb + (size_t)ln * L_ + jb + 32, 0, 1);

    v8f s0, s1;
#pragma unroll
    for (int r = 0; r < 8; r++) { s0[r] = 0.f; s1[r] = 0.f; }
    v16bf bk;
    bk = ldfrag(kb + (size_t)(jb + ln) * HD_ + half * 8);
    s0 = wmma_bf16(aq0, bk, s0);
    bk = ldfrag(kb + (size_t)(jb + 16 + ln) * HD_ + half * 8);
    s1 = wmma_bf16(aq0, bk, s1);
    bk = ldfrag(kb + (size_t)(jb + ln) * HD_ + 32 + half * 8);
    s0 = wmma_bf16(aq1, bk, s0);
    bk = ldfrag(kb + (size_t)(jb + 16 + ln) * HD_ + 32 + half * 8);
    s1 = wmma_bf16(aq1, bk, s1);

    // causal mask (key > query -> -inf)
#pragma unroll
    for (int r = 0; r < 8; r++) {
      int qrow = q0 + r + 8 * half;
      if (jb + ln      > qrow) s0[r] = -1e30f;
      if (jb + 16 + ln > qrow) s1[r] = -1e30f;
    }

    // online softmax update + stage P (bf16) to LDS in A-fragment order
#pragma unroll
    for (int r = 0; r < 8; r++) {
      float rowmax = rmax16(fmaxf(s0[r], s1[r]));
      float mn = fmaxf(m[r], rowmax);
      float alpha = __expf(m[r] - mn);
      m[r] = mn;
      float e0 = __expf(s0[r] - mn);
      float e1 = __expf(s1[r] - mn);
      lsum[r] = lsum[r] * alpha + rsum16(e0 + e1);
#pragma unroll
      for (int nt = 0; nt < 4; nt++) o[nt][r] *= alpha;
      int prow = r + 8 * half;
      lds_p[prow * 32 + ln]      = f2bf(e0);
      lds_p[prow * 32 + 16 + ln] = f2bf(e1);
    }
    __syncthreads();
    v16bf ap = ldfrag(&lds_p[ln * 32 + half * 8]);
#pragma unroll
    for (int nt = 0; nt < 4; nt++) {
      v16bf bv = ldfrag(vb + (size_t)(nt * 16 + ln) * L_ + jb + half * 8);
      o[nt] = wmma_bf16(ap, bv, o[nt]);
    }
    __syncthreads();
  }

  // normalize and write (B*L, D) bf16 for the wo GEMM
#pragma unroll
  for (int nt = 0; nt < 4; nt++)
#pragma unroll
    for (int r = 0; r < 8; r++) {
      int qrow = q0 + r + 8 * half;
      float val = o[nt][r] / lsum[r];
      obf[(size_t)(b * L_ + qrow) * D_ + h * HD_ + nt * 16 + ln] = f2bf(val);
    }
}

// ---------------------------------------------------------------------------
// Router second matvec + sigmoid: probs[row] = sigmoid(dot(r1[row], w2) + b2)
// ---------------------------------------------------------------------------
__global__ __launch_bounds__(256) void router2_kernel(const unsigned short* __restrict__ r1,
                                                      const float* __restrict__ w2,
                                                      const float* __restrict__ b2,
                                                      float* __restrict__ probs) {
  __shared__ float red[256];
  const int row = blockIdx.x, tid = threadIdx.x;
  float s = 0.f;
#pragma unroll
  for (int i = 0; i < 4; i++) {
    int c = tid + i * 256;
    s += bf2f(r1[(size_t)row * D_ + c]) * w2[c];
  }
  red[tid] = s;
  __syncthreads();
  for (int off = 128; off > 0; off >>= 1) {
    if (tid < off) red[tid] += red[tid + off];
    __syncthreads();
  }
  if (tid == 0) {
    float lg = red[0] + b2[0];
    probs[row] = 1.f / (1.f + __expf(-lg));
  }
}

// Serial per-batch scan: hard mask (forced 1 at t=0), counts, and
// boundary_positions = stable desc argsort of hard (ones then zeros, in order)
__global__ void scan_kernel(const float* __restrict__ probs, int* __restrict__ hard,
                            int* __restrict__ cnts, int* __restrict__ bp_out,
                            int* __restrict__ cnt_out) {
  if (threadIdx.x != 0) return;
  const int b = blockIdx.x;
  const float* pr = probs + b * L_;
  int* hd = hard + b * L_;
  int c = 0;
  for (int l = 0; l < L_; l++) {
    int hv = (l == 0) ? 1 : (pr[l] > 0.5f ? 1 : 0);
    hd[l] = hv;
    c += hv;
  }
  cnts[b] = c;
  cnt_out[b] = c;
  int idx = 0;
  for (int l = 0; l < L_; l++) if (hd[l])  bp_out[b * L_ + idx++] = l;
  for (int l = 0; l < L_; l++) if (!hd[l]) bp_out[b * L_ + idx++] = l;
}

__global__ void avg_kernel(const int* __restrict__ cnts, float* __restrict__ out) {
  int s = 0;
  for (int b = 0; b < B_; b++) s += cnts[b];
  *out = (float)L_ * (float)B_ / (float)s;
}

// compressed[b,s,:] = x[b, bp[s], :] for s < counts (bm==hard forward), else 0
__global__ __launch_bounds__(256) void compress_kernel(const float* __restrict__ x,
                                                       const int* __restrict__ bp,
                                                       const int* __restrict__ cnts,
                                                       float* __restrict__ comp) {
  const int b = blockIdx.y;
  const int d = blockIdx.x * 256 + threadIdx.x;
  const int c = cnts[b];
  for (int s = 0; s < L_; s++) {
    float v = 0.f;
    if (s < c) v = x[(size_t)(b * L_ + bp[b * L_ + s]) * D_ + d];
    comp[(size_t)(b * L_ + s) * D_ + d] = v;
  }
}

// ---------------------------------------------------------------------------
// Host orchestration
// ---------------------------------------------------------------------------
extern "C" void kernel_launch(void* const* d_in, const int* in_sizes, int n_in,
                              void* d_out, int out_size, void* d_ws, size_t ws_size,
                              hipStream_t stream) {
  (void)in_sizes; (void)n_in; (void)out_size; (void)ws_size;
  const float* x_in    = (const float*)d_in[0];
  const float* cosp    = (const float*)d_in[1];
  const float* sinp    = (const float*)d_in[2];
  const float* wq      = (const float*)d_in[3];
  const float* wk      = (const float*)d_in[4];
  const float* wv      = (const float*)d_in[5];
  const float* wo      = (const float*)d_in[6];
  const float* w_gate  = (const float*)d_in[7];
  const float* w_up    = (const float*)d_in[8];
  const float* w_down  = (const float*)d_in[9];
  const float* norm1   = (const float*)d_in[10];
  const float* norm2   = (const float*)d_in[11];
  const float* finalw  = (const float*)d_in[12];
  const float* r_w1    = (const float*)d_in[13];
  const float* r_b1    = (const float*)d_in[14];
  const float* r_w2    = (const float*)d_in[15];
  const float* r_b2    = (const float*)d_in[16];

  // workspace carve-out (~104 MB)
  char* p = (char*)d_ws;
  float*          x_cur   = (float*)p;          p += (size_t)M_ * D_ * sizeof(float);   // 16MB residual stream
  unsigned short* h_bf    = (unsigned short*)p; p += (size_t)M_ * D_ * 2;               //  8MB norm/bf16 act
  float*          scr     = (float*)p;          p += (size_t)M_ * D_ * sizeof(float);   // 16MB q/k/v f32 scratch
  unsigned short* q_bf    = (unsigned short*)p; p += (size_t)M_ * D_ * 2;               //  8MB [b,h,l,hd]
  unsigned short* k_bf    = (unsigned short*)p; p += (size_t)M_ * D_ * 2;               //  8MB [b,h,l,hd]
  unsigned short* vT_bf   = (unsigned short*)p; p += (size_t)M_ * D_ * 2;               //  8MB [b,h,hd,l]
  unsigned short* attn_bf = (unsigned short*)p; p += (size_t)M_ * D_ * 2;               //  8MB attn out / xfin
  unsigned short* gate_bf = (unsigned short*)p; p += (size_t)M_ * HID_ * 2;             // 32MB gate/act
  float*          probs   = (float*)p;          p += (size_t)M_ * sizeof(float);
  int*            hard    = (int*)p;            p += (size_t)M_ * sizeof(int);
  int*            cnts    = (int*)p;            p += 256;

  // output layout: [x][compressed][boundary(int32)][counts(int32)][avg]
  float* out_x    = (float*)d_out;
  float* out_comp = out_x + (size_t)M_ * D_;
  int*   out_bp   = (int*)(out_comp + (size_t)M_ * D_);
  int*   out_cnt  = out_bp + M_;
  float* out_avg  = (float*)(out_cnt + B_);

  const dim3 gDD(M_ / 128, D_ / 128);     // (32, 8)
  const dim3 gDH(M_ / 128, HID_ / 128);   // (32, 32)

  copy_f32<<<(M_ * D_ / 4 + 255) / 256, 256, 0, stream>>>(x_in, x_cur, M_ * D_ / 4);

  for (int i = 0; i < NL_; i++) {
    const float* wq_i = wq + (size_t)i * D_ * D_;
    const float* wk_i = wk + (size_t)i * D_ * D_;
    const float* wv_i = wv + (size_t)i * D_ * D_;
    const float* wo_i = wo + (size_t)i * D_ * D_;
    const float* wg_i = w_gate + (size_t)i * D_ * HID_;
    const float* wu_i = w_up   + (size_t)i * D_ * HID_;
    const float* wd_i = w_down + (size_t)i * HID_ * D_;

    rmsnorm_kernel<false><<<M_, 256, 0, stream>>>(x_cur, norm1 + i * D_, h_bf, nullptr);

    gemm_kernel<0><<<gDD, 256, 0, stream>>>(h_bf, wq_i, D_, D_, scr, nullptr, nullptr, nullptr);
    rope_pack_kernel<<<(B_ * L_ * H_ * 32) / 256, 256, 0, stream>>>(scr, cosp, sinp, q_bf, 0.125f);
    gemm_kernel<0><<<gDD, 256, 0, stream>>>(h_bf, wk_i, D_, D_, scr, nullptr, nullptr, nullptr);
    rope_pack_kernel<<<(B_ * L_ * H_ * 32) / 256, 256, 0, stream>>>(scr, cosp, sinp, k_bf, 1.0f);
    gemm_kernel<0><<<gDD, 256, 0, stream>>>(h_bf, wv_i, D_, D_, scr, nullptr, nullptr, nullptr);
    vtrans_kernel<<<(M_ * D_) / 256, 256, 0, stream>>>(scr, vT_bf);

    attn_kernel<<<dim3(L_ / 16, B_ * H_), 32, 0, stream>>>(q_bf, k_bf, vT_bf, attn_bf);

    gemm_kernel<1><<<gDD, 256, 0, stream>>>(attn_bf, wo_i, D_, D_, x_cur, nullptr, x_cur, nullptr);

    rmsnorm_kernel<false><<<M_, 256, 0, stream>>>(x_cur, norm2 + i * D_, h_bf, nullptr);
    gemm_kernel<3><<<gDH, 256, 0, stream>>>(h_bf, wg_i, HID_, D_, nullptr, gate_bf, nullptr, nullptr);
    gemm_kernel<4><<<gDH, 256, 0, stream>>>(h_bf, wu_i, HID_, D_, nullptr, gate_bf, nullptr, nullptr);
    gemm_kernel<1><<<gDD, 256, 0, stream>>>(gate_bf, wd_i, D_, HID_, x_cur, nullptr, x_cur, nullptr);
  }

  // final norm -> d_out (f32) + bf16 copy for router
  rmsnorm_kernel<true><<<M_, 256, 0, stream>>>(x_cur, finalw, attn_bf, out_x);

  // router: silu(x @ r_w1 + r_b1) -> bf16, then matvec + sigmoid
  gemm_kernel<5><<<gDD, 256, 0, stream>>>(attn_bf, r_w1, D_, D_, nullptr, h_bf, nullptr, r_b1);
  router2_kernel<<<M_, 256, 0, stream>>>(h_bf, r_w2, r_b2, probs);

  // chunking: scan + boundary argsort + counts, avg, gather-compress
  scan_kernel<<<B_, 32, 0, stream>>>(probs, hard, cnts, out_bp, out_cnt);
  avg_kernel<<<1, 1, 0, stream>>>(cnts, out_avg);
  compress_kernel<<<dim3(D_ / 256, B_), 256, 0, stream>>>(out_x, out_bp, cnts, out_comp);
}
